// TemporalGraphCooperativePolicy_37314675868476
// MI455X (gfx1250) — matrix operand order repaired and checked by reading, not compiled, not run on hardware
//
#include <hip/hip_runtime.h>
#include <hip/hip_bf16.h>
#include <cstdint>

typedef _Float16 half_t;
typedef __attribute__((ext_vector_type(16))) _Float16 v16h;
typedef __attribute__((ext_vector_type(8)))  float    v8f;

#define N_NODES 4096
#define F_IN    64
#define H_DIM   128
#define C_CAND  64
#define CFEAT   10
#define NEGV    (-1000000000.0f)
#define KSTEP   64
#define BPAD    72   // padded half-stride (144B = 9x16B: aligned + bank-spread)

__device__ __forceinline__ int lane_id() { return (int)(threadIdx.x & 31); }

// ---------------- CDNA5 async global->LDS (ASYNCcnt-tracked) ------------------
__device__ __forceinline__ void async_ld_b128(void* lds_ptr, const void* gptr) {
  uint32_t lds = (uint32_t)(uintptr_t)lds_ptr;
  uint64_t ga = (uint64_t)(uintptr_t)gptr;
  asm volatile("global_load_async_to_lds_b128 %0, %1, off"
               :: "v"(lds), "v"(ga) : "memory");
}

template <int N>
__device__ __forceinline__ void wait_async() {
  asm volatile("s_wait_asynccnt %0" :: "n"(N) : "memory");
}

// ---------------- WMMA fragment helpers (v_wmma_f32_16x16x32_f16) -------------
template <typename F>
__device__ __forceinline__ v16h make_a_frag(F get) {
  int l = lane_id();
  int m = l & 15;
  int hi = (l >> 4) & 1;
  v16h a;
#pragma unroll
  for (int v = 0; v < 8; ++v) {
    int kb = (v < 4) ? (hi * 8 + 2 * v) : (16 + hi * 8 + 2 * (v - 4));
    a[2 * v]     = (half_t)get(m, kb);
    a[2 * v + 1] = (half_t)get(m, kb + 1);
  }
  return a;
}

template <typename F>
__device__ __forceinline__ v16h make_b_frag(F get) {
  int l = lane_id();
  int n = l & 15;
  int hi = (l >> 4) & 1;
  v16h b;
#pragma unroll
  for (int v = 0; v < 8; ++v) {
    int k = (v < 4) ? (hi * 8 + 2 * v) : (16 + hi * 8 + 2 * (v - 4));
    b[2 * v]     = (half_t)get(k, n);
    b[2 * v + 1] = (half_t)get(k + 1, n);
  }
  return b;
}

template <typename F>
__device__ __forceinline__ void emit_c(const v8f& c, F put) {
  int l = lane_id();
  int n = l & 15;
  int mb = (l >> 4) ? 8 : 0;
#pragma unroll
  for (int r = 0; r < 8; ++r) put(mb + r, n, c[r]);
}

__device__ __forceinline__ v8f wmma_f16(v16h a, v16h b, v8f c) {
  return __builtin_amdgcn_wmma_f32_16x16x32_f16(false, a, false, b, (short)0, c,
                                                false, false);
}

__device__ __forceinline__ float gelu_erf(float x) {
  return 0.5f * x * (1.0f + erff(x * 0.70710678118654752440f));
}

// ------------------------------------------------------------------------------
// C[M,128] = A[M,K] @ B[N,K]^T (+bias). 1 wave per 16x16 tile.
// OUTH_MODE: 0=none, 1=row-major [M,128], 2=transposed [128,N_NODES]
template <bool BIAS, bool OUTF, int OUTH_MODE>
__global__ __launch_bounds__(32)
void k_gemm_xWT(const float* __restrict__ A, const float* __restrict__ B,
                const float* __restrict__ bias, float* __restrict__ outF,
                half_t* __restrict__ outH, int K) {
  int mb = blockIdx.x * 16;
  int nb = blockIdx.y * 16;
  v8f c = {};
  for (int k0 = 0; k0 < K; k0 += 32) {
    v16h a = make_a_frag([&](int m, int k) { return A[(mb + m) * K + k0 + k]; });
    v16h b = make_b_frag([&](int k, int n) { return B[(nb + n) * K + k0 + k]; });
    c = wmma_f16(a, b, c);
  }
  float bv = BIAS ? bias[nb + (lane_id() & 15)] : 0.0f;
  emit_c(c, [&](int m, int n, float v) {
    float x = v + bv;
    if (OUTF) outF[(mb + m) * H_DIM + nb + n] = x;
    if (OUTH_MODE == 1) outH[(mb + m) * H_DIM + nb + n] = (half_t)x;
    if (OUTH_MODE == 2) outH[(size_t)(nb + n) * N_NODES + mb + m] = (half_t)x;
  });
}

__global__ void k_svec(const float* __restrict__ h, const float* __restrict__ as,
                       const float* __restrict__ ad, float* __restrict__ ssrc,
                       float* __restrict__ sdst) {
  int i = blockIdx.x * blockDim.x + threadIdx.x;
  if (i >= N_NODES) return;
  float s1 = 0.f, s2 = 0.f;
#pragma unroll 4
  for (int k = 0; k < H_DIM; ++k) {
    float v = h[i * H_DIM + k];
    s1 += v * as[k];
    s2 += v * ad[k];
  }
  ssrc[i] = s1;
  sdst[i] = s2;
}

__global__ __launch_bounds__(256)
void k_rowstats(const int* __restrict__ adj, const float* __restrict__ ssrc,
                const float* __restrict__ sdst, float* __restrict__ mrow,
                float* __restrict__ zinv) {
  __shared__ float ebuf[N_NODES];
  __shared__ float red[256];
  int i = blockIdx.x;
  int t = threadIdx.x;
  float si = ssrc[i];
  float mx = -3.4e38f;
  for (int j = t; j < N_NODES; j += 256) {
    float v = si + sdst[j];
    v = v > 0.f ? v : 0.2f * v;
    v = (adj[i * N_NODES + j] > 0) ? v : NEGV;
    ebuf[j] = v;
    mx = fmaxf(mx, v);
  }
  red[t] = mx;
  __syncthreads();
  for (int s = 128; s > 0; s >>= 1) {
    if (t < s) red[t] = fmaxf(red[t], red[t + s]);
    __syncthreads();
  }
  float m = red[0];
  float sum = 0.f;
  for (int j = t; j < N_NODES; j += 256) sum += __expf(ebuf[j] - m);
  __syncthreads();
  red[t] = sum;
  __syncthreads();
  for (int s = 128; s > 0; s >>= 1) {
    if (t < s) red[t] += red[t + s];
    __syncthreads();
  }
  if (t == 0) {
    mrow[i] = m;
    zinv[i] = 1.0f / red[0];
  }
}

// out[i,:] = act( sum_j softmax(e_ij) h[j,:] ).  16 rows x 128 cols per block,
// 8 waves (one 16-col tile each). h held TRANSPOSED in global (h16T[H][N]) so
// both A and B WMMA fragments are k-contiguous b128 LDS reads. Adjacency and
// h-slabs stream via global_load_async_to_lds_b128, double-buffered with a
// counted s_wait_asynccnt so the next 20KB slab DMA overlaps current compute.
template <bool RESID>
__global__ __launch_bounds__(256)
void k_attn_agg(const int* __restrict__ adj, const float* __restrict__ ssrc,
                const float* __restrict__ sdst, const float* __restrict__ mrow,
                const float* __restrict__ zinv, const half_t* __restrict__ h16T,
                const float* __restrict__ resid, float* __restrict__ outF) {
  __shared__ half_t BtileT[2][H_DIM][BPAD];  // 2 x 18KB (col-major slab, padded)
  __shared__ int    AdjT[2][16][KSTEP];      // 2 x 4KB
  __shared__ half_t Atile[16][BPAD];         // 2.25KB
  __shared__ float sm[16], sz[16], ss[16];
  int rb = blockIdx.x * 16;
  int tid = threadIdx.x;
  if (tid < 16) {
    sm[tid] = mrow[rb + tid];
    sz[tid] = zinv[rb + tid];
    ss[tid] = ssrc[rb + tid];
  }
  // 5 async b128 issues per wave per slab: 4 for h-columns, 1 for adjacency
  auto stage = [&](int buf, int kb) {
#pragma unroll
    for (int u = 0; u < 4; ++u) {
      int idx = tid + u * 256;            // 0..1023
      int col = idx >> 3, part = idx & 7; // col 0..127, 16B part 0..7
      async_ld_b128(&BtileT[buf][col][part * 8],
                    h16T + (size_t)col * N_NODES + kb + part * 8);
    }
    int r = tid >> 4, g = tid & 15;       // 16 rows x 16 chunks of 4 ints
    async_ld_b128(&AdjT[buf][r][g * 4],
                  adj + (size_t)(rb + r) * N_NODES + kb + g * 4);
  };
  stage(0, 0);
  __syncthreads();
  int nb = (tid >> 5) * 16;
  v8f c = {};
  for (int it = 0; it < N_NODES / KSTEP; ++it) {
    int kb = it * KSTEP;
    int cur = it & 1;
    if (it + 1 < N_NODES / KSTEP) {
      stage(cur ^ 1, kb + KSTEP);
      wait_async<5>();  // current slab landed; next slab still in flight
    } else {
      wait_async<0>();
    }
    __syncthreads();
    // probability A-tile for this slab (16x64, 4 entries/thread), adj from LDS
#pragma unroll
    for (int u = 0; u < 4; ++u) {
      int idx = tid + u * 256;
      int r = idx >> 6, jj = idx & 63;
      float v = ss[r] + sdst[kb + jj];
      v = v > 0.f ? v : 0.2f * v;
      v = (AdjT[cur][r][jj] > 0) ? v : NEGV;
      Atile[r][jj] = (half_t)(__expf(v - sm[r]) * sz[r]);
    }
    __syncthreads();
#pragma unroll
    for (int kk = 0; kk < KSTEP; kk += 32) {
      v16h a = make_a_frag([&](int m, int k) { return (float)Atile[m][kk + k]; });
      v16h b = make_b_frag(
          [&](int k, int n) { return (float)BtileT[cur][nb + n][kk + k]; });
      c = wmma_f16(a, b, c);
    }
    __syncthreads();
  }
  emit_c(c, [&](int m, int n, float v) {
    float x = fmaxf(v, 0.0f);
    if (RESID) x += resid[(rb + m) * H_DIM + nb + n];
    outF[(rb + m) * H_DIM + nb + n] = x;
  });
}

__global__ __launch_bounds__(128)
void k_ln(const float* __restrict__ h, const float* __restrict__ w,
          const float* __restrict__ b, float* __restrict__ out) {
  __shared__ float red[128];
  int i = blockIdx.x;
  int t = threadIdx.x;
  float x = h[i * H_DIM + t];
  red[t] = x;
  __syncthreads();
  for (int s = 64; s > 0; s >>= 1) {
    if (t < s) red[t] += red[t + s];
    __syncthreads();
  }
  float mu = red[0] * (1.0f / H_DIM);
  __syncthreads();
  float d = x - mu;
  red[t] = d * d;
  __syncthreads();
  for (int s = 64; s > 0; s >>= 1) {
    if (t < s) red[t] += red[t + s];
    __syncthreads();
  }
  float var = red[0] * (1.0f / H_DIM);
  out[i * H_DIM + t] = d * rsqrtf(var + 1e-5f) * w[t] + b[t];
}

// Fused per-node candidate encoder + scorer. Block = node, 256 threads.
__global__ __launch_bounds__(256)
void k_policy(const float* __restrict__ cfAll, const float* __restrict__ amAll,
              const half_t* __restrict__ q16, const float* __restrict__ Wc1,
              const float* __restrict__ bc1, const float* __restrict__ Wc2,
              const float* __restrict__ bc2, const float* __restrict__ Ws1,
              const float* __restrict__ bs1, const float* __restrict__ Ws2,
              const float* __restrict__ bs2, float* __restrict__ out) {
  __shared__ __align__(16) unsigned char smem[48 * 1024];
  half_t(*fusedS)[256] = (half_t(*)[256])smem;              // [0,32K) ph3-4
  half_t(*cfpad)[32] = (half_t(*)[32])smem;                 // [0,4K)  ph1
  half_t(*cand1)[128] = (half_t(*)[128])(smem + 16 * 1024); // [16K,32K) ph1-2
  half_t(*cand2)[128] = (half_t(*)[128])(smem + 32 * 1024); // [32K,48K) ph2-3
  __shared__ float logitsAcc[8][64];
  __shared__ half_t qh[128];
  __shared__ float amsum;

  int node = blockIdx.x;
  int tid = threadIdx.x;
  int wv = tid >> 5;
  int nb = wv * 16;
  const float* cf = cfAll + (size_t)node * C_CAND * CFEAT;

  if (tid < 128) qh[tid] = q16[node * H_DIM + tid];
  if (tid == 0) {
    float s = 0.f;
    for (int c = 0; c < C_CAND; ++c) s += amAll[node * C_CAND + c];
    amsum = s;
  }
  for (int idx = tid; idx < 64 * 32; idx += 256) {
    int r = idx >> 5, k = idx & 31;
    cfpad[r][k] = (half_t)(k < CFEAT ? cf[r * CFEAT + k] : 0.0f);
  }
  __syncthreads();
  // cand1 = gelu(cf @ Wc1^T + bc1), K=32 (padded); weight frag shared over mt
  {
    v16h bw = make_b_frag([&](int k, int n) {
      return k < CFEAT ? Wc1[(nb + n) * CFEAT + k] : 0.0f;
    });
    for (int mt = 0; mt < 4; ++mt) {
      v16h a =
          make_a_frag([&](int m, int k) { return (float)cfpad[mt * 16 + m][k]; });
      v8f c = {};
      c = wmma_f16(a, bw, c);
      emit_c(c, [&](int m, int n, float v) {
        cand1[mt * 16 + m][nb + n] = (half_t)gelu_erf(v + bc1[nb + n]);
      });
    }
  }
  __syncthreads();
  // cand2 = gelu(cand1 @ Wc2^T + bc2), K=128; 4 weight frags hoisted
  {
    v16h bw[4];
#pragma unroll
    for (int ki = 0; ki < 4; ++ki)
      bw[ki] = make_b_frag(
          [&](int k, int n) { return Wc2[(nb + n) * H_DIM + ki * 32 + k]; });
    for (int mt = 0; mt < 4; ++mt) {
      v8f c = {};
#pragma unroll
      for (int ki = 0; ki < 4; ++ki) {
        v16h a = make_a_frag(
            [&](int m, int k) { return (float)cand1[mt * 16 + m][ki * 32 + k]; });
        c = wmma_f16(a, bw[ki], c);
      }
      emit_c(c, [&](int m, int n, float v) {
        cand2[mt * 16 + m][nb + n] = (half_t)gelu_erf(v + bc2[nb + n]);
      });
    }
  }
  __syncthreads();
  // fused = [q*cand2, cand2]  [64,256]
  for (int idx = tid; idx < 64 * 256; idx += 256) {
    int r = idx >> 8, cx = idx & 255;
    float v = (cx < 128) ? (float)qh[cx] * (float)cand2[r][cx]
                         : (float)cand2[r][cx - 128];
    fusedS[r][cx] = (half_t)v;
  }
  __syncthreads();
  // s = gelu(fused @ Ws1^T + bs1); logits partial = s . Ws2; 8 frags hoisted
  {
    int n = lane_id() & 15;
    int mbase = (lane_id() >> 4) ? 8 : 0;
    int col = nb + n;
    float w2 = Ws2[col];
    float b1 = bs1[col];
    v16h bw[8];
#pragma unroll
    for (int ki = 0; ki < 8; ++ki)
      bw[ki] = make_b_frag(
          [&](int k, int nn) { return Ws1[(nb + nn) * 256 + ki * 32 + k]; });
    for (int mt = 0; mt < 4; ++mt) {
      v8f c = {};
#pragma unroll
      for (int ki = 0; ki < 8; ++ki) {
        v16h a = make_a_frag(
            [&](int m, int k) { return (float)fusedS[mt * 16 + m][ki * 32 + k]; });
        c = wmma_f16(a, bw[ki], c);
      }
#pragma unroll
      for (int r = 0; r < 8; ++r) {
        float g = gelu_erf(c[r] + b1) * w2;
        for (int off = 8; off >= 1; off >>= 1) g += __shfl_xor(g, off, 32);
        if (n == 0) logitsAcc[wv][mt * 16 + mbase + r] = g;
      }
    }
  }
  __syncthreads();
  if (tid < C_CAND) {
    int c = tid;
    float lg = bs2[0];
#pragma unroll
    for (int w = 0; w < 8; ++w) lg += logitsAcc[w][c];
    const float* r = cf + c * CFEAT;
    lg += 20.0f * r[0] + 4.0f * r[4] + 1.5f * r[5] + 1.5f * r[1] - 1.5f * r[2] +
          1.2f * r[6] + 2.5f * r[7] + 1.6f * r[8] + 1.2f * r[9];
    float a = amAll[node * C_CAND + c];
    if (c == 0 && amsum <= 0.0f) a = 1.0f;
    out[node * C_CAND + c] = (a > 0.0f) ? lg : NEGV;
  }
}

// ------------------------------------------------------------------------------
extern "C" void kernel_launch(void* const* d_in, const int* in_sizes, int n_in,
                              void* d_out, int out_size, void* d_ws,
                              size_t ws_size, hipStream_t stream) {
  (void)in_sizes; (void)n_in; (void)out_size; (void)ws_size;
  const float* x   = (const float*)d_in[0];
  const float* cf  = (const float*)d_in[1];
  const int*   adj = (const int*)d_in[2];
  const float* am  = (const float*)d_in[3];
  const float* W1  = (const float*)d_in[4];
  const float* a1s = (const float*)d_in[5];
  const float* a1d = (const float*)d_in[6];
  const float* W2  = (const float*)d_in[7];
  const float* a2s = (const float*)d_in[8];
  const float* a2d = (const float*)d_in[9];
  const float* lnw = (const float*)d_in[10];
  const float* lnb = (const float*)d_in[11];
  const float* Wc1 = (const float*)d_in[12];
  const float* bc1 = (const float*)d_in[13];
  const float* Wc2 = (const float*)d_in[14];
  const float* bc2 = (const float*)d_in[15];
  const float* Wq  = (const float*)d_in[16];
  const float* bq  = (const float*)d_in[17];
  const float* Ws1 = (const float*)d_in[18];
  const float* bs1 = (const float*)d_in[19];
  const float* Ws2 = (const float*)d_in[20];
  const float* bs2 = (const float*)d_in[21];
  float* out = (float*)d_out;

  char* p = (char*)d_ws;
  float*  hpre   = (float*)p;  p += (size_t)N_NODES * H_DIM * 4;  // pre-act / h2
  half_t* h16T   = (half_t*)p; p += (size_t)N_NODES * H_DIM * 2;  // transposed
  float*  h1     = (float*)p;  p += (size_t)N_NODES * H_DIM * 4;
  float*  hln    = (float*)p;  p += (size_t)N_NODES * H_DIM * 4;
  half_t* q16v   = (half_t*)p; p += (size_t)N_NODES * H_DIM * 2;
  float*  ssrc   = (float*)p;  p += (size_t)N_NODES * 4;
  float*  sdst   = (float*)p;  p += (size_t)N_NODES * 4;
  float*  mrow   = (float*)p;  p += (size_t)N_NODES * 4;
  float*  zinv   = (float*)p;  p += (size_t)N_NODES * 4;

  dim3 gGemm(N_NODES / 16, H_DIM / 16);

  // ---- GAT layer 1 ----
  k_gemm_xWT<false, true, 2>
      <<<gGemm, 32, 0, stream>>>(x, W1, nullptr, hpre, h16T, F_IN);
  k_svec<<<N_NODES / 256, 256, 0, stream>>>(hpre, a1s, a1d, ssrc, sdst);
  k_rowstats<<<N_NODES, 256, 0, stream>>>(adj, ssrc, sdst, mrow, zinv);
  k_attn_agg<false><<<N_NODES / 16, 256, 0, stream>>>(adj, ssrc, sdst, mrow,
                                                      zinv, h16T, nullptr, h1);
  // ---- GAT layer 2 (residual) ----
  k_gemm_xWT<false, true, 2>
      <<<gGemm, 32, 0, stream>>>(h1, W2, nullptr, hpre, h16T, H_DIM);
  k_svec<<<N_NODES / 256, 256, 0, stream>>>(hpre, a2s, a2d, ssrc, sdst);
  k_rowstats<<<N_NODES, 256, 0, stream>>>(adj, ssrc, sdst, mrow, zinv);
  k_attn_agg<true><<<N_NODES / 16, 256, 0, stream>>>(adj, ssrc, sdst, mrow,
                                                     zinv, h16T, h1, hpre);
  // ---- LayerNorm + query projection ----
  k_ln<<<N_NODES, 128, 0, stream>>>(hpre, lnw, lnb, hln);
  k_gemm_xWT<true, false, 1>
      <<<gGemm, 32, 0, stream>>>(hln, Wq, bq, nullptr, q16v, H_DIM);
  // ---- fused candidate encoder + scorer ----
  k_policy<<<N_NODES, 256, 0, stream>>>(cf, am, q16v, Wc1, bc1, Wc2, bc2, Ws1,
                                        bs1, Ws2, bs2, out);
}
